// GraphSAGEWithNorm_82205674045441
// MI455X (gfx1250) — matrix operand order repaired and checked by reading, not compiled, not run on hardware
//
#include <hip/hip_runtime.h>

#define N_NODES 100000
#define E_EDGES 1600000
#define IN_DIM 128
#define HID 128
#define OUT_DIM 64
#define BN_EPS 1e-5f

typedef __bf16 bf16;
typedef __attribute__((ext_vector_type(16))) __bf16 v16bf;
typedef __attribute__((ext_vector_type(8)))  __bf16 v8bf;
typedef __attribute__((ext_vector_type(4)))  __bf16 v4bf;
typedef __attribute__((ext_vector_type(8)))  float  v8f;

// ---------------------------------------------------------------------------
// Degree count: cnt[dst] += 1 per edge (f32 hardware atomics, L2-resident)
// ---------------------------------------------------------------------------
__global__ void deg_kernel(const int* __restrict__ dst, float* __restrict__ cnt) {
    int e = blockIdx.x * blockDim.x + threadIdx.x;
    if (e < E_EDGES) unsafeAtomicAdd(&cnt[dst[e]], 1.0f);
}

// ---------------------------------------------------------------------------
// Layer-1 message scatter: one wave per edge, float4 gather + 4 f32 atomics
// per lane. x table (51.2MB) and s table (51.2MB) both fit in 192MB L2.
// ---------------------------------------------------------------------------
__global__ void scatter_f32_kernel(const int* __restrict__ src,
                                   const int* __restrict__ dst,
                                   const float* __restrict__ x,
                                   float* __restrict__ s) {
    long long gid = (long long)blockIdx.x * blockDim.x + threadIdx.x;
    int e    = (int)(gid >> 5);
    int lane = (int)(gid & 31);
    if (e >= E_EDGES) return;
    int sn = src[e], dn = dst[e];
    const float4 v = *(const float4*)(x + (size_t)sn * IN_DIM + lane * 4);
    float* p = s + (size_t)dn * IN_DIM + lane * 4;
    unsafeAtomicAdd(p + 0, v.x);
    unsafeAtomicAdd(p + 1, v.y);
    unsafeAtomicAdd(p + 2, v.z);
    unsafeAtomicAdd(p + 3, v.w);
}

// Layer-2 scatter: gather bf16 hidden features, accumulate f32 atomics.
__global__ void scatter_bf16_kernel(const int* __restrict__ src,
                                    const int* __restrict__ dst,
                                    const bf16* __restrict__ h,
                                    float* __restrict__ s) {
    long long gid = (long long)blockIdx.x * blockDim.x + threadIdx.x;
    int e    = (int)(gid >> 5);
    int lane = (int)(gid & 31);
    if (e >= E_EDGES) return;
    int sn = src[e], dn = dst[e];
    v4bf v = *(const v4bf*)(h + (size_t)sn * HID + lane * 4);
    float* p = s + (size_t)dn * HID + lane * 4;
    unsafeAtomicAdd(p + 0, (float)v[0]);
    unsafeAtomicAdd(p + 1, (float)v[1]);
    unsafeAtomicAdd(p + 2, (float)v[2]);
    unsafeAtomicAdd(p + 3, (float)v[3]);
}

// ---------------------------------------------------------------------------
// Mean-normalize + convert to bf16 (fold the row scale before the GEMM:
// diag(1/cnt) * (s @ W^T) == (s/cnt) @ W^T). Also converts x once.
// ---------------------------------------------------------------------------
__global__ void convert1_kernel(const float* __restrict__ s,
                                const float* __restrict__ cnt,
                                const float* __restrict__ x,
                                bf16* __restrict__ agg_bf,
                                bf16* __restrict__ x_bf) {
    long long i = (long long)blockIdx.x * blockDim.x + threadIdx.x;
    if (i >= (long long)N_NODES * HID) return;
    int node = (int)(i >> 7);
    float inv = 1.0f / fmaxf(cnt[node], 1.0f);
    agg_bf[i] = (bf16)(s[i] * inv);
    x_bf[i]   = (bf16)x[i];
}

__global__ void convert2_kernel(const float* __restrict__ s,
                                const float* __restrict__ cnt,
                                bf16* __restrict__ agg_bf) {
    long long i = (long long)blockIdx.x * blockDim.x + threadIdx.x;
    if (i >= (long long)N_NODES * HID) return;
    int node = (int)(i >> 7);
    float inv = 1.0f / fmaxf(cnt[node], 1.0f);
    agg_bf[i] = (bf16)(s[i] * inv);
}

// ---------------------------------------------------------------------------
// Pack weight W[rows x 128] into per-lane WMMA B fragments (bf16).
// B layout (16-bit, 32x16): lane L holds column N=L%16; element j holds
// K = (j&7) + 8*(L>>4) + 16*(j>>3) (same K order as the A fragment).
// packed[((nt*4 + kb)*32 + L)*16 + j] = W[nt*16 + L%16, kb*32 + K(L,j)]
// -> each wave's B fragment is one contiguous 32-byte load per lane.
// ---------------------------------------------------------------------------
__global__ void pack_w_kernel(const float* __restrict__ W,
                              bf16* __restrict__ P, int rows) {
    int p = blockIdx.x * blockDim.x + threadIdx.x;
    if (p >= rows * 128) return;
    int j  = p & 15;
    int L  = (p >> 4) & 31;
    int kb = (p >> 9) & 3;
    int nt = p >> 11;
    P[p] = (bf16)W[(nt * 16 + (L & 15)) * 128 + (kb * 32 + (j & 7) + ((L >> 4) << 3) + ((j >> 3) << 4))];
}

// A fragment (16-bit, 16x32): lane L holds row M=L%16; lanes 0-15 carry
// K in {0..7, 16..23}, lanes 16-31 carry K in {8..15, 24..31}. Two
// contiguous 16B runs per lane from row-major bf16.
__device__ inline v16bf load_a_frag(const bf16* __restrict__ base,
                                    int row, int kb, int lane) {
    const bf16* p = base + (size_t)row * 128 + kb * 32 + (lane >> 4) * 8;
    v8bf lo = *(const v8bf*)p;
    v8bf hi = *(const v8bf*)(p + 16);
    v16bf a;
#pragma unroll
    for (int i = 0; i < 8; ++i) { a[i] = lo[i]; a[i + 8] = hi[i]; }
    return a;
}

__device__ inline v16bf load_b_frag(const bf16* __restrict__ packed,
                                    int nt, int kb, int lane) {
    return *(const v16bf*)(packed + (((size_t)nt * 4 + kb) * 32 + lane) * 16);
}

__device__ inline v8f wmma_bf16(v16bf a, v16bf b, v8f c) {
    return __builtin_amdgcn_wmma_f32_16x16x32_bf16(
        false, a, false, b, (short)0, c, false, false);
}

// ---------------------------------------------------------------------------
// Layer 1: h[16 rows x 128] per block; 8 waves, wave w = 16-column N-tile.
// h = agg @ W1_l^T + b1_l + x @ W1_r^T   (K=128 -> 4 k-blocks, 8 WMMA/wave)
// ---------------------------------------------------------------------------
__global__ void __launch_bounds__(256) gemm1_kernel(
    const bf16* __restrict__ agg, const bf16* __restrict__ xb,
    const bf16* __restrict__ Pl, const bf16* __restrict__ Pr,
    const float* __restrict__ bias, float* __restrict__ h) {
    int wave = threadIdx.x >> 5;   // N-tile 0..7
    int lane = threadIdx.x & 31;
    int tile = blockIdx.x;         // 6250 row tiles
    int row  = tile * 16 + (lane & 15);
    v8f acc = {};
#pragma unroll
    for (int kb = 0; kb < 4; ++kb) {
        acc = wmma_bf16(load_a_frag(agg, row, kb, lane), load_b_frag(Pl, wave, kb, lane), acc);
        acc = wmma_bf16(load_a_frag(xb,  row, kb, lane), load_b_frag(Pr, wave, kb, lane), acc);
    }
    int col = wave * 16 + (lane & 15);
    float bcol = bias[col];
    int mbase = (lane < 16) ? 0 : 8;
#pragma unroll
    for (int r = 0; r < 8; ++r)
        h[(size_t)(tile * 16 + mbase + r) * HID + col] = acc[r] + bcol;
}

// Layer 2: out[32 rows x 64] per block; 8 waves = 2 row-tiles x 4 N-tiles.
__global__ void __launch_bounds__(256) gemm2_kernel(
    const bf16* __restrict__ agg, const bf16* __restrict__ hb,
    const bf16* __restrict__ Pl, const bf16* __restrict__ Pr,
    const float* __restrict__ bias, float* __restrict__ out) {
    int wave = threadIdx.x >> 5;
    int lane = threadIdx.x & 31;
    int nt   = wave & 3;
    int tile = blockIdx.x * 2 + (wave >> 2);   // 3125 blocks
    int row  = tile * 16 + (lane & 15);
    v8f acc = {};
#pragma unroll
    for (int kb = 0; kb < 4; ++kb) {
        acc = wmma_bf16(load_a_frag(agg, row, kb, lane), load_b_frag(Pl, nt, kb, lane), acc);
        acc = wmma_bf16(load_a_frag(hb,  row, kb, lane), load_b_frag(Pr, nt, kb, lane), acc);
    }
    int col = nt * 16 + (lane & 15);
    float bcol = bias[col];
    int mbase = (lane < 16) ? 0 : 8;
#pragma unroll
    for (int r = 0; r < 8; ++r)
        out[(size_t)(tile * 16 + mbase + r) * OUT_DIM + col] = acc[r] + bcol;
}

// ---------------------------------------------------------------------------
// BatchNorm stats: thread = feature, block-strided rows, partial sums ->
// f32 atomics into accum[0..127]=sum, accum[128..255]=sumsq.
// ---------------------------------------------------------------------------
__global__ void bn_stats_kernel(const float* __restrict__ h,
                                float* __restrict__ accum) {
    int f = threadIdx.x;
    float s = 0.f, s2 = 0.f;
    for (int row = blockIdx.x; row < N_NODES; row += gridDim.x) {
        float v = h[(size_t)row * HID + f];
        s += v; s2 += v * v;
    }
    unsafeAtomicAdd(&accum[f], s);
    unsafeAtomicAdd(&accum[HID + f], s2);
}

// Fused BN (training-mode, biased var) + ReLU + bf16 store for layer 2.
__global__ void bn_apply_kernel(const float* __restrict__ h,
                                const float* __restrict__ accum,
                                const float* __restrict__ gamma,
                                const float* __restrict__ beta,
                                bf16* __restrict__ hb) {
    long long i = (long long)blockIdx.x * blockDim.x + threadIdx.x;
    if (i >= (long long)N_NODES * HID) return;
    int f = (int)(i & (HID - 1));
    const float invN = 1.0f / (float)N_NODES;
    float mean = accum[f] * invN;
    float var  = accum[HID + f] * invN - mean * mean;
    float sc   = gamma[f] * rsqrtf(var + BN_EPS);
    float v    = (h[i] - mean) * sc + beta[f];
    hb[i] = (bf16)fmaxf(v, 0.0f);
}

// ---------------------------------------------------------------------------
// log_softmax over 64 cols: one wave32 per row, lane handles col and col+32,
// shuffle tree for max and sum-of-exps.
// ---------------------------------------------------------------------------
__global__ void logsoftmax_kernel(float* __restrict__ out) {
    int wid  = (int)(((long long)blockIdx.x * blockDim.x + threadIdx.x) >> 5);
    int lane = threadIdx.x & 31;
    if (wid >= N_NODES) return;
    float* row = out + (size_t)wid * OUT_DIM;
    float a = row[lane], b = row[lane + 32];
    float m = fmaxf(a, b);
#pragma unroll
    for (int off = 16; off > 0; off >>= 1) m = fmaxf(m, __shfl_xor(m, off, 32));
    float e = __expf(a - m) + __expf(b - m);
#pragma unroll
    for (int off = 16; off > 0; off >>= 1) e += __shfl_xor(e, off, 32);
    float lse = m + __logf(e);
    row[lane]      = a - lse;
    row[lane + 32] = b - lse;
}

// ---------------------------------------------------------------------------
extern "C" void kernel_launch(void* const* d_in, const int* in_sizes, int n_in,
                              void* d_out, int out_size, void* d_ws, size_t ws_size,
                              hipStream_t stream) {
    const float* x     = (const float*)d_in[0];
    const int*   ei    = (const int*)d_in[1];   // [2, E]; src row then dst row
    const float* W1_l  = (const float*)d_in[2];
    const float* b1_l  = (const float*)d_in[3];
    const float* W1_r  = (const float*)d_in[4];
    const float* gamma = (const float*)d_in[5];
    const float* beta  = (const float*)d_in[6];
    const float* W2_l  = (const float*)d_in[7];
    const float* b2_l  = (const float*)d_in[8];
    const float* W2_r  = (const float*)d_in[9];
    float* out = (float*)d_out;
    const int* src = ei;
    const int* dst = ei + E_EDGES;

    // Workspace carve-up with reuse:
    //   bufA (f32 N*128): s1 -> h_pre -> s2
    //   bfA  (bf16 N*128): agg1 -> agg2
    //   bfB  (bf16 N*128): x_bf -> h1_bf (post BN/ReLU)
    char* ws = (char*)d_ws;
    size_t off = 0;
    auto carve = [&](size_t bytes) {
        size_t o = off; off += (bytes + 255) & ~(size_t)255; return o;
    };
    float* bufA  = (float*)(ws + carve((size_t)N_NODES * HID * 4));
    bf16*  bfA   = (bf16*) (ws + carve((size_t)N_NODES * HID * 2));
    bf16*  bfB   = (bf16*) (ws + carve((size_t)N_NODES * HID * 2));
    float* cnt   = (float*)(ws + carve((size_t)N_NODES * 4));
    float* accum = (float*)(ws + carve(2 * HID * 4));
    bf16*  P1l   = (bf16*) (ws + carve((size_t)HID * IN_DIM * 2));
    bf16*  P1r   = (bf16*) (ws + carve((size_t)HID * IN_DIM * 2));
    bf16*  P2l   = (bf16*) (ws + carve((size_t)OUT_DIM * HID * 2));
    bf16*  P2r   = (bf16*) (ws + carve((size_t)OUT_DIM * HID * 2));

    hipMemsetAsync(bufA,  0, (size_t)N_NODES * HID * 4, stream);
    hipMemsetAsync(cnt,   0, (size_t)N_NODES * 4, stream);
    hipMemsetAsync(accum, 0, 2 * HID * 4, stream);

    pack_w_kernel<<<(HID * IN_DIM + 255) / 256, 256, 0, stream>>>(W1_l, P1l, HID);
    pack_w_kernel<<<(HID * IN_DIM + 255) / 256, 256, 0, stream>>>(W1_r, P1r, HID);
    pack_w_kernel<<<(OUT_DIM * HID + 255) / 256, 256, 0, stream>>>(W2_l, P2l, OUT_DIM);
    pack_w_kernel<<<(OUT_DIM * HID + 255) / 256, 256, 0, stream>>>(W2_r, P2r, OUT_DIM);

    // ---- Layer 1 ----
    deg_kernel<<<(E_EDGES + 255) / 256, 256, 0, stream>>>(dst, cnt);
    scatter_f32_kernel<<<E_EDGES * 32 / 256, 256, 0, stream>>>(src, dst, x, bufA);
    convert1_kernel<<<(N_NODES * HID) / 256, 256, 0, stream>>>(bufA, cnt, x, bfA, bfB);
    gemm1_kernel<<<N_NODES / 16, 256, 0, stream>>>(bfA, bfB, P1l, P1r, b1_l, bufA);

    // ---- BN + ReLU ----
    bn_stats_kernel<<<512, 128, 0, stream>>>(bufA, accum);
    bn_apply_kernel<<<(N_NODES * HID) / 256, 256, 0, stream>>>(bufA, accum, gamma, beta, bfB);

    // ---- Layer 2 ----
    hipMemsetAsync(bufA, 0, (size_t)N_NODES * HID * 4, stream);
    scatter_bf16_kernel<<<E_EDGES * 32 / 256, 256, 0, stream>>>(src, dst, bfB, bufA);
    convert2_kernel<<<(N_NODES * HID) / 256, 256, 0, stream>>>(bufA, cnt, bfA);
    gemm2_kernel<<<N_NODES / 32, 256, 0, stream>>>(bfA, bfB, P2l, P2r, b2_l, out);

    // ---- log_softmax (in place on d_out) ----
    logsoftmax_kernel<<<N_NODES * 32 / 256, 256, 0, stream>>>(out);
}
